// ACE_41979010351671
// MI455X (gfx1250) — compile-verified, matrix operand order checked by reading, not run on hardware
//
#include <hip/hip_runtime.h>
#include <hip/hip_bf16.h>
#include <math.h>

#define BATCH 8
#define HH 64
#define WW 64
#define HWX 4096
#define C2C 1024
#define SCALE_ATT 0.17677669529663687f  // 32^-0.5

typedef __attribute__((ext_vector_type(16))) __bf16        v16bf;
typedef __attribute__((ext_vector_type(8)))  float         v8f;
typedef __attribute__((ext_vector_type(4)))  unsigned int  u32x4;

union FragAB { unsigned int u[8]; v16bf v; u32x4 q[2]; };
union FragC  { float f[8]; v8f v; };
struct Slot { FragAB a0, a1, b0, b1, b2, b3; };

__device__ __forceinline__ unsigned short f2bf(float f){
  unsigned int u = __float_as_uint(f);
  u += 0x7FFFu + ((u >> 16) & 1u);
  return (unsigned short)(u >> 16);
}
__device__ __forceinline__ unsigned int packbf(float a, float b){
  return (unsigned int)f2bf(a) | ((unsigned int)f2bf(b) << 16);
}
__device__ __forceinline__ v8f wmma_bf16(v16bf a, v16bf b, v8f c){
  return __builtin_amdgcn_wmma_f32_16x16x32_bf16(false, a, false, b, (short)0, c, false, false);
}
// Swizzled tile layout: u32x4 buffer; block index = (kc*Rows + r); each block = 4 u32x4
// (16 packed bf16-pair dwords = one 32-deep K chunk for one row/column r).
__device__ __forceinline__ void load_frag(FragAB& f, const u32x4* base, size_t blk, int half){
  f.q[0] = base[blk*4 + half];
  f.q[1] = base[blk*4 + 2 + half];
}
__device__ __forceinline__ void zero_frag(FragAB& f){
#pragma unroll
  for(int i=0;i<8;i++) f.u[i] = 0u;
}
__device__ __forceinline__ void slot_compute(FragC (&acc)[2][4], const Slot& s){
  acc[0][0].v = wmma_bf16(s.a0.v, s.b0.v, acc[0][0].v);
  acc[1][0].v = wmma_bf16(s.a1.v, s.b0.v, acc[1][0].v);
  acc[0][1].v = wmma_bf16(s.a0.v, s.b1.v, acc[0][1].v);
  acc[1][1].v = wmma_bf16(s.a1.v, s.b1.v, acc[1][1].v);
  acc[0][2].v = wmma_bf16(s.a0.v, s.b2.v, acc[0][2].v);
  acc[1][2].v = wmma_bf16(s.a1.v, s.b2.v, acc[1][2].v);
  acc[0][3].v = wmma_bf16(s.a0.v, s.b3.v, acc[0][3].v);
  acc[1][3].v = wmma_bf16(s.a1.v, s.b3.v, acc[1][3].v);
}

// ---------------- generic 1x1-conv GEMM: Out[b][m][n] = act(sum_k W[m][k]*In[b][k][n])
// grid: (N/64, M/32, B), block 32; 32M x 64N per wave; static double-buffered pipeline
// mode 0: y = acc*s[m]+b[m];  mode 1: y = other[b][m][n] * sigmoid(acc*s[m]+b[m])
__global__ void __launch_bounds__(32, 1)
ACE_gemm1x1(const u32x4* __restrict__ Wp,   // swizzled [KC][M][16dw]
            const u32x4* __restrict__ Inp,  // swizzled [B][KC][N][16dw]
            float* __restrict__ Out,
            const float* __restrict__ scale,
            const float* __restrict__ bias,
            const float* __restrict__ other,
            int M, int K, int N, int mode){
  int lane = threadIdx.x, half = lane >> 4, nl = lane & 15;
  int n0 = blockIdx.x * 64;
  int m0 = blockIdx.y * 32;
  int bb = blockIdx.z;
  int KC = K >> 5;                               // even for all uses
  const u32x4* inb = Inp + (size_t)bb * KC * N * 4;
  FragC acc[2][4];
#pragma unroll
  for(int mi=0;mi<2;mi++)
#pragma unroll
    for(int t=0;t<4;t++)
#pragma unroll
      for(int i=0;i<8;i++) acc[mi][t].f[i] = 0.f;

  auto loadSlot = [&](Slot& s, int kc){
    load_frag(s.a0, Wp, (size_t)kc*M + m0      + nl, half);
    load_frag(s.a1, Wp, (size_t)kc*M + m0 + 16 + nl, half);
    size_t nb = (size_t)kc*N + n0 + nl;
    load_frag(s.b0, inb, nb,      half);
    load_frag(s.b1, inb, nb + 16, half);
    load_frag(s.b2, inb, nb + 32, half);
    load_frag(s.b3, inb, nb + 48, half);
  };

  Slot sA, sB;
  loadSlot(sA, 0);
  for(int kc = 0; kc < KC; kc += 2){
    loadSlot(sB, kc + 1);
    slot_compute(acc, sA);
    if(kc + 2 < KC) loadSlot(sA, kc + 2);
    slot_compute(acc, sB);
  }
  float* outb = Out + (size_t)bb * M * N;
  const float* othb = other ? other + (size_t)bb * M * N : nullptr;
#pragma unroll
  for(int mi=0;mi<2;mi++)
#pragma unroll
    for(int t=0;t<4;t++){
      int n = n0 + t*16 + nl;
#pragma unroll
      for(int i=0;i<8;i++){
        int m = m0 + mi*16 + i + 8*half;
        float v = acc[mi][t].f[i] * scale[m] + bias[m];
        if(mode == 1) v = othb[(size_t)m * N + n] / (1.f + __expf(-v));
        outb[(size_t)m * N + n] = v;
      }
    }
}

// ---------------- vertical stripe conv (7x1): grid (H, Cout/32, B), block 32
__global__ void __launch_bounds__(32, 1)
ACE_stripe_v(const u32x4* __restrict__ Wp,    // swizzled over K=7*Cin
             const u32x4* __restrict__ Inp,   // swizzled [B][Cin/32][HWX][16dw]
             float* __restrict__ Out,
             int Cout, int Cin, int accumulate){
  int lane = threadIdx.x, half = lane >> 4, nl = lane & 15;
  int h  = blockIdx.x;
  int m0 = blockIdx.y * 32;
  int bb = blockIdx.z;
  int cpt = Cin >> 5;            // chunks per tap
  int KC = 7 * cpt;              // even (14 or 224)
  const u32x4* inb = Inp + (size_t)bb * cpt * HWX * 4;
  FragC acc[2][4];
#pragma unroll
  for(int mi=0;mi<2;mi++)
#pragma unroll
    for(int t=0;t<4;t++)
#pragma unroll
      for(int i=0;i<8;i++) acc[mi][t].f[i] = 0.f;

  auto loadSlot = [&](Slot& s, int kc){
    load_frag(s.a0, Wp, (size_t)kc*Cout + m0      + nl, half);
    load_frag(s.a1, Wp, (size_t)kc*Cout + m0 + 16 + nl, half);
    int tap = kc / cpt, cik = kc - tap*cpt;
    int hs = h + tap - 3;
    if(hs >= 0 && hs < HH){
      size_t nb = (size_t)cik*HWX + hs*WW + nl;
      load_frag(s.b0, inb, nb,      half);
      load_frag(s.b1, inb, nb + 16, half);
      load_frag(s.b2, inb, nb + 32, half);
      load_frag(s.b3, inb, nb + 48, half);
    } else {
      zero_frag(s.b0); zero_frag(s.b1); zero_frag(s.b2); zero_frag(s.b3);
    }
  };

  Slot sA, sB;
  loadSlot(sA, 0);
  for(int kc = 0; kc < KC; kc += 2){
    loadSlot(sB, kc + 1);
    slot_compute(acc, sA);
    if(kc + 2 < KC) loadSlot(sA, kc + 2);
    slot_compute(acc, sB);
  }
  float* outb = Out + (size_t)bb * Cout * HWX;
#pragma unroll
  for(int mi=0;mi<2;mi++)
#pragma unroll
    for(int t=0;t<4;t++){
      int w = t*16 + nl;
#pragma unroll
      for(int i=0;i<8;i++){
        int m = m0 + mi*16 + i + 8*half;
        size_t idx = (size_t)m * HWX + h*WW + w;
        float v = acc[mi][t].f[i];
        if(accumulate) outb[idx] += v; else outb[idx] = v;
      }
    }
}

// ---------------- horizontal stripe conv (1x7)
__global__ void __launch_bounds__(32, 1)
ACE_stripe_h(const u32x4* __restrict__ Wp,
             const u32x4* __restrict__ Inp,
             float* __restrict__ Out,
             int Cout, int Cin, int accumulate){
  int lane = threadIdx.x, half = lane >> 4, nl = lane & 15;
  int h  = blockIdx.x;
  int m0 = blockIdx.y * 32;
  int bb = blockIdx.z;
  int cpt = Cin >> 5;
  int KC = 7 * cpt;
  const u32x4* inb = Inp + (size_t)bb * cpt * HWX * 4;
  FragC acc[2][4];
#pragma unroll
  for(int mi=0;mi<2;mi++)
#pragma unroll
    for(int t=0;t<4;t++)
#pragma unroll
      for(int i=0;i<8;i++) acc[mi][t].f[i] = 0.f;

  auto ld1 = [&](FragAB& f, int cik, int ws){
    if(ws >= 0 && ws < WW) load_frag(f, inb, (size_t)cik*HWX + h*WW + ws, half);
    else zero_frag(f);
  };
  auto loadSlot = [&](Slot& s, int kc){
    load_frag(s.a0, Wp, (size_t)kc*Cout + m0      + nl, half);
    load_frag(s.a1, Wp, (size_t)kc*Cout + m0 + 16 + nl, half);
    int tap = kc / cpt, cik = kc - tap*cpt;
    int wb = nl + tap - 3;
    ld1(s.b0, cik, wb);
    ld1(s.b1, cik, wb + 16);
    ld1(s.b2, cik, wb + 32);
    ld1(s.b3, cik, wb + 48);
  };

  Slot sA, sB;
  loadSlot(sA, 0);
  for(int kc = 0; kc < KC; kc += 2){
    loadSlot(sB, kc + 1);
    slot_compute(acc, sA);
    if(kc + 2 < KC) loadSlot(sA, kc + 2);
    slot_compute(acc, sB);
  }
  float* outb = Out + (size_t)bb * Cout * HWX;
#pragma unroll
  for(int mi=0;mi<2;mi++)
#pragma unroll
    for(int t=0;t<4;t++){
      int w = t*16 + nl;
#pragma unroll
      for(int i=0;i<8;i++){
        int m = m0 + mi*16 + i + 8*half;
        size_t idx = (size_t)m * HWX + h*WW + w;
        float v = acc[mi][t].f[i];
        if(accumulate) outb[idx] += v; else outb[idx] = v;
      }
    }
}

// ---------------- activation swizzle-pack: f32 [B][K][N] -> swizzled bf16-pair dwords
__global__ void ACE_pack2s(const float* __restrict__ src, unsigned int* __restrict__ dst,
                           const float* __restrict__ cs, const float* __restrict__ cb,
                           int K, int KC, int N, int total){
  int o = blockIdx.x * blockDim.x + threadIdx.x;
  if(o >= total) return;
  int q = o & 15;
  int n = (o >> 4) % N;
  int rr = o / (16 * N);         // b*KC + kc
  int kc = rr % KC;
  int b  = rr / KC;
  int k = kc*32 + 2*q;
  size_t sbase = ((size_t)b * K + k) * N + n;
  float v0 = src[sbase], v1 = src[sbase + N];
  if(cs){ v0 = v0*cs[k] + cb[k]; v1 = v1*cs[k+1] + cb[k+1]; }
  dst[o] = packbf(v0, v1);
}

// ---------------- generic weight swizzle-pack: f32 row-major [M][K] -> [KC][M][16dw]
__global__ void ACE_pack_w(const float* __restrict__ src, unsigned int* __restrict__ dst,
                           int M, int K, int total){
  int o = blockIdx.x * blockDim.x + threadIdx.x;
  if(o >= total) return;
  int q = o & 15;
  int m = (o >> 4) % M;
  int kc = o / (16 * M);
  int k = kc*32 + 2*q;
  dst[o] = packbf(src[(size_t)m*K + k], src[(size_t)m*K + k + 1]);
}

// fused q/k/v weights -> swizzled [KC=8][1024][16dw] + fused scale/bias
__global__ void ACE_pack_qkv(const float* wq, const float* sq, const float* bq,
                             const float* wk, const float* sk, const float* bk,
                             const float* wv, const float* sv, const float* bv,
                             unsigned int* W, float* so, float* bo){
  int o = blockIdx.x * blockDim.x + threadIdx.x;
  if(o >= 1024*128) return;
  int q = o & 15;
  int m = (o >> 4) & 1023;
  int kc = o >> 14;
  int k = kc*32 + 2*q;
  const float* wsrc; const float* ssrc; const float* bsrc; int mr;
  if(m < 256){ wsrc = wq; ssrc = sq; bsrc = bq; mr = m; }
  else if(m < 512){ wsrc = wk; ssrc = sk; bsrc = bk; mr = m - 256; }
  else { wsrc = wv; ssrc = sv; bsrc = bv; mr = m - 512; }
  W[o] = packbf(wsrc[mr*256 + k], wsrc[mr*256 + k + 1]);
  if(o < 1024*16 && q == 0){ so[m] = ssrc[mr]; bo[m] = bsrc[mr]; }
}

// kv (64,1024,7) -> swizzled A: M=64, K=tap*1024+ci (KC=224)
__global__ void ACE_pack_stripeA(const float* kv, unsigned int* dst){
  int o = blockIdx.x * blockDim.x + threadIdx.x;
  if(o >= 64*3584) return;
  int q = o & 15;
  int m = (o >> 4) & 63;
  int kc = o >> 10;
  int k = kc*32 + 2*q;
  int tap = k >> 10, ci = k & 1023;
  dst[o] = packbf(kv[(m*1024 + ci)*7 + tap], kv[(m*1024 + ci + 1)*7 + tap]);
}
// kv -> swizzled transposed weights: M=1024 (ci), K=tap*64+cm (KC=14)
__global__ void ACE_pack_stripeB(const float* kv, unsigned int* dst){
  int o = blockIdx.x * blockDim.x + threadIdx.x;
  if(o >= 1024*224) return;
  int q = o & 15;
  int m = (o >> 4) & 1023;
  int kc = o >> 14;
  int k = kc*32 + 2*q;
  int tap = k >> 6, cm = k & 63;
  dst[o] = packbf(kv[(cm*1024 + m)*7 + tap], kv[((cm+1)*1024 + m)*7 + tap]);
}

// ---------------- depthwise 3x3 + BN + ReLU over qkv0 -> branch (f32)
__global__ void ACE_dwconv(const float* __restrict__ in, const float* __restrict__ wdw,
                           const float* __restrict__ s, const float* __restrict__ b,
                           float* __restrict__ out){
  int idx = blockIdx.x * blockDim.x + threadIdx.x;
  if(idx >= BATCH*C2C*HWX) return;
  int hw = idx & 4095, c = (idx >> 12) & 1023, bb = idx >> 22;
  int h = hw >> 6, w = hw & 63;
  const float* pin = in + ((size_t)bb*C2C + c) * HWX;
  float a = 0.f;
#pragma unroll
  for(int dy=0; dy<3; dy++){
    int hs = h + dy - 1;
    if(hs < 0 || hs >= HH) continue;
#pragma unroll
    for(int dx=0; dx<3; dx++){
      int wsd = w + dx - 1;
      if(wsd < 0 || wsd >= WW) continue;
      a += wdw[c*9 + dy*3 + dx] * pin[hs*WW + wsd];
    }
  }
  float v = a * s[c] + b[c];
  out[idx] = fmaxf(v, 0.f);
}

// ---------------- act_dn stats: per-(b,c) spatial max & sumexp
__global__ void ACE_actdn_stats(const float* __restrict__ x, float* __restrict__ stats){
  int bc = blockIdx.x;
  int tid = threadIdx.x;
  const float* p = x + (size_t)bc * HWX;
  __shared__ float red[256];
  float mx = -1e30f;
  for(int i = tid; i < HWX; i += 256) mx = fmaxf(mx, p[i]);
  red[tid] = mx; __syncthreads();
  for(int o = 128; o > 0; o >>= 1){ if(tid < o) red[tid] = fmaxf(red[tid], red[tid+o]); __syncthreads(); }
  mx = red[0]; __syncthreads();
  float se = 0.f;
  for(int i = tid; i < HWX; i += 256) se += __expf(p[i] - mx);
  red[tid] = se; __syncthreads();
  for(int o = 128; o > 0; o >>= 1){ if(tid < o) red[tid] += red[tid+o]; __syncthreads(); }
  if(tid == 0){ stats[bc*2] = mx; stats[bc*2+1] = red[0]; }
}

// act_dn apply -> swizzled packed output [B][KC=2][HWX][16dw], vector writes
__global__ void ACE_actdn_apply(const float* __restrict__ x, const float* __restrict__ stats,
                                unsigned int* __restrict__ y){
  int idx = blockIdx.x * blockDim.x + threadIdx.x;
  if(idx >= BATCH*8*HWX) return;
  int s = idx & 4095, head = (idx >> 12) & 7, bb = idx >> 15;
  float vals[8]; float den = 0.f;
#pragma unroll
  for(int cg=0; cg<8; cg++){
    int bc = bb*64 + head*8 + cg;
    float v = __expf(x[(size_t)bc*HWX + s] - stats[bc*2]) / stats[bc*2+1];
    vals[cg] = v; den += v;
  }
  float inv = 1.f / (den + 1e-6f);
  int kp0 = head*4;
  int kc = kp0 >> 4, q0 = kp0 & 15;
  u32x4 pk;
  pk.x = packbf(vals[0]*inv, vals[1]*inv);
  pk.y = packbf(vals[2]*inv, vals[3]*inv);
  pk.z = packbf(vals[4]*inv, vals[5]*inv);
  pk.w = packbf(vals[6]*inv, vals[7]*inv);
  u32x4* dst = (u32x4*)y;
  dst[(((size_t)(bb*2 + kc)*HWX + s)*16 + q0) >> 2] = pk;
}

// ---------------- means over W / over H for all 1024 channels of qkv0
__global__ void ACE_mean_w(const float* __restrict__ in, float* __restrict__ out){
  int idx = blockIdx.x * blockDim.x + threadIdx.x;
  if(idx >= BATCH*C2C*HH) return;
  int h = idx & 63, c = (idx >> 6) & 1023, bb = idx >> 16;
  const float* p = in + ((size_t)bb*C2C + c) * HWX + h*WW;
  float s = 0.f;
  for(int w=0; w<WW; w++) s += p[w];
  out[idx] = s * (1.f/64.f);
}
__global__ void ACE_mean_h(const float* __restrict__ in, float* __restrict__ out){
  int idx = blockIdx.x * blockDim.x + threadIdx.x;
  if(idx >= BATCH*C2C*WW) return;
  int w = idx & 63, c = (idx >> 6) & 1023, bb = idx >> 16;
  const float* p = in + ((size_t)bb*C2C + c) * HWX + w;
  float s = 0.f;
  for(int h=0; h<HH; h++) s += p[h*WW];
  out[idx] = s * (1.f/64.f);
}

// ---------------- row/col attention per (b,head); writes relu(out) swizzled-packed (K=512,N=64)
__global__ void ACE_attn_axis(const float* __restrict__ mean,
                              const float* __restrict__ peq,
                              const float* __restrict__ pek,
                              unsigned int* __restrict__ outp){
  __shared__ float qs[2048], ks[2048], vs[4096], at[4096];
  int bb = blockIdx.x >> 3, head = blockIdx.x & 7;
  int tid = threadIdx.x;
  const float* mb = mean + (size_t)bb * C2C * 64;
  for(int idx = tid; idx < 2048; idx += 64){
    int c = idx >> 6, i = idx & 63;
    float coord = (i + 0.5f)*0.25f - 0.5f;
    coord = fminf(fmaxf(coord, 0.f), 15.f);
    int lo = (int)floorf(coord);
    int hi = lo + 1 < 16 ? lo + 1 : 15;
    float t = coord - (float)lo;
    int cg = head*32 + c;
    qs[idx] = mb[cg*64 + i]       + peq[cg*16+lo]*(1.f-t) + peq[cg*16+hi]*t;
    ks[idx] = mb[(256+cg)*64 + i] + pek[cg*16+lo]*(1.f-t) + pek[cg*16+hi]*t;
  }
  for(int idx = tid; idx < 4096; idx += 64){
    int d = idx >> 6, j = idx & 63;
    vs[idx] = mb[(512 + head*64 + d)*64 + j];
  }
  __syncthreads();
  int i = tid;
  float mx = -1e30f;
  for(int j=0; j<64; j++){
    float s = 0.f;
    for(int c=0; c<32; c++) s += qs[c*64+i] * ks[c*64+j];
    s *= SCALE_ATT;
    at[i*64+j] = s; mx = fmaxf(mx, s);
  }
  float se = 0.f;
  for(int j=0; j<64; j++){ float e = __expf(at[i*64+j]-mx); at[i*64+j] = e; se += e; }
  float inv = 1.f / se;
  unsigned int* ob = outp + (size_t)bb * 256 * 64;
  for(int d=0; d<64; d+=2){
    float o0 = 0.f, o1 = 0.f;
    for(int j=0; j<64; j++){ o0 += at[i*64+j]*vs[d*64+j]; o1 += at[i*64+j]*vs[(d+1)*64+j]; }
    int kp = head*32 + (d>>1);
    int kc = kp >> 4, q = kp & 15;
    ob[((size_t)kc*64 + i)*16 + q] = packbf(fmaxf(o0*inv,0.f), fmaxf(o1*inv,0.f));
  }
}

// ---------------- vsum = relu(v + row_bcast + col_bcast), swizzled packed [B][KC=16][HWX][16dw]
__global__ void ACE_vsum(const float* __restrict__ qkv0, const float* __restrict__ rconv,
                         const float* __restrict__ cconv, unsigned int* __restrict__ out){
  int o = blockIdx.x * blockDim.x + threadIdx.x;
  if(o >= BATCH*16*HWX*16) return;
  int q = o & 15;
  int hw = (o >> 4) & 4095;
  int kc = (o >> 16) & 15;
  int bb = o >> 20;
  int c = 2*(kc*16 + q);
  int h = hw >> 6, w = hw & 63;
  const float* qb  = qkv0 + ((size_t)bb*C2C + 512 + c) * HWX + hw;
  const float* rb  = rconv + ((size_t)bb*512 + c) * 64;
  const float* cbp = cconv + ((size_t)bb*512 + c) * 64;
  float v0 = fmaxf(qb[0]   + rb[h]    + cbp[w],    0.f);
  float v1 = fmaxf(qb[HWX] + rb[64+h] + cbp[64+w], 0.f);
  out[o] = packbf(v0, v1);
}

extern "C" void kernel_launch(void* const* d_in, const int* in_sizes, int n_in,
                              void* d_out, int out_size, void* d_ws, size_t ws_size,
                              hipStream_t stream) {
  (void)in_sizes; (void)n_in; (void)out_size; (void)ws_size;
  const float* x      = (const float*)d_in[0];
  const float* wq     = (const float*)d_in[1];
  const float* sq     = (const float*)d_in[2];
  const float* bq     = (const float*)d_in[3];
  const float* wk     = (const float*)d_in[4];
  const float* sk     = (const float*)d_in[5];
  const float* bk     = (const float*)d_in[6];
  const float* wv     = (const float*)d_in[7];
  const float* sv     = (const float*)d_in[8];
  const float* bv     = (const float*)d_in[9];
  const float* w_dw   = (const float*)d_in[10];
  const float* s_dw   = (const float*)d_in[11];
  const float* b_dw   = (const float*)d_in[12];
  const float* cov_s  = (const float*)d_in[13];
  const float* cov_b  = (const float*)d_in[14];
  const float* kv     = (const float*)d_in[15];
  const float* kv3    = (const float*)d_in[16];
  const float* w_pw   = (const float*)d_in[17];
  const float* s_pw   = (const float*)d_in[18];
  const float* b_pw   = (const float*)d_in[19];
  const float* w_row  = (const float*)d_in[20];
  const float* s_row  = (const float*)d_in[21];
  const float* b_row  = (const float*)d_in[22];
  const float* w_col  = (const float*)d_in[23];
  const float* s_col  = (const float*)d_in[24];
  const float* b_col  = (const float*)d_in[25];
  const float* w_proj = (const float*)d_in[26];
  const float* s_proj = (const float*)d_in[27];
  const float* b_proj = (const float*)d_in[28];
  const float* pe_rowq= (const float*)d_in[29];
  const float* pe_rowk= (const float*)d_in[30];
  const float* pe_colq= (const float*)d_in[31];
  const float* pe_colk= (const float*)d_in[32];
  float* out = (float*)d_out;

  char* cur = (char*)d_ws;
  auto alloc = [&](size_t bytes)->char*{
    char* r = cur; cur += (bytes + 255) & ~(size_t)255; return r;
  };
  float* qkv0    = (float*)alloc((size_t)BATCH*C2C*HWX*4);
  float* branch  = (float*)alloc((size_t)BATCH*C2C*HWX*4);
  unsigned int* branchp = (unsigned int*)alloc((size_t)BATCH*512*HWX*4);
  unsigned int* xnp     = (unsigned int*)alloc((size_t)BATCH*512*HWX*4);
  unsigned int* xp      = (unsigned int*)alloc((size_t)BATCH*128*HWX*4);
  float* x1      = (float*)alloc((size_t)BATCH*64*HWX*4);
  float* x3      = (float*)alloc((size_t)BATCH*64*HWX*4);
  unsigned int* x1sp = (unsigned int*)alloc((size_t)BATCH*32*HWX*4);
  unsigned int* x3sp = (unsigned int*)alloc((size_t)BATCH*32*HWX*4);
  float* qkv_pw  = (float*)alloc((size_t)BATCH*256*HWX*4);
  unsigned int* vsump = (unsigned int*)alloc((size_t)BATCH*256*HWX*4);
  float* meanW   = (float*)alloc((size_t)BATCH*C2C*64*4);
  float* meanH   = (float*)alloc((size_t)BATCH*C2C*64*4);
  unsigned int* out_rp = (unsigned int*)alloc((size_t)BATCH*256*64*4);
  unsigned int* out_cp = (unsigned int*)alloc((size_t)BATCH*256*64*4);
  float* out_rc  = (float*)alloc((size_t)BATCH*512*64*4);
  float* out_cc  = (float*)alloc((size_t)BATCH*512*64*4);
  float* stats1  = (float*)alloc(512*2*4);
  float* stats3  = (float*)alloc(512*2*4);
  unsigned int* Wqkv  = (unsigned int*)alloc(1024*128*4);
  float* sqkv = (float*)alloc(1024*4);
  float* bqkv = (float*)alloc(1024*4);
  unsigned int* kvA   = (unsigned int*)alloc((size_t)64*3584*4);
  unsigned int* kvB   = (unsigned int*)alloc((size_t)1024*224*4);
  unsigned int* kv3A  = (unsigned int*)alloc((size_t)64*3584*4);
  unsigned int* kv3B  = (unsigned int*)alloc((size_t)1024*224*4);
  unsigned int* Wpw   = (unsigned int*)alloc(256*512*4);
  unsigned int* Wrow  = (unsigned int*)alloc(512*256*4);
  unsigned int* Wcol  = (unsigned int*)alloc(512*256*4);
  unsigned int* Wproj = (unsigned int*)alloc(256*256*4);

  // 1. pack weights into swizzled bf16 fragment layout
  ACE_pack_qkv<<<(1024*128+255)/256, 256, 0, stream>>>(wq,sq,bq, wk,sk,bk, wv,sv,bv, Wqkv, sqkv, bqkv);
  ACE_pack_stripeA<<<(64*3584+255)/256, 256, 0, stream>>>(kv,  kvA);
  ACE_pack_stripeB<<<(1024*224+255)/256, 256, 0, stream>>>(kv,  kvB);
  ACE_pack_stripeA<<<(64*3584+255)/256, 256, 0, stream>>>(kv3, kv3A);
  ACE_pack_stripeB<<<(1024*224+255)/256, 256, 0, stream>>>(kv3, kv3B);
  ACE_pack_w<<<(256*512+255)/256, 256, 0, stream>>>(w_pw,   Wpw,   256, 1024, 256*512);
  ACE_pack_w<<<(512*256+255)/256, 256, 0, stream>>>(w_row,  Wrow,  512, 512,  512*256);
  ACE_pack_w<<<(512*256+255)/256, 256, 0, stream>>>(w_col,  Wcol,  512, 512,  512*256);
  ACE_pack_w<<<(256*256+255)/256, 256, 0, stream>>>(w_proj, Wproj, 256, 512,  256*256);

  // 2. swizzle-pack x, then fused q/k/v 1x1 conv + BN (WMMA)
  {
    int total = BATCH*128*HWX;
    ACE_pack2s<<<(total+255)/256, 256, 0, stream>>>(x, xp, nullptr, nullptr, 256, 8, HWX, total);
  }
  ACE_gemm1x1<<<dim3(HWX/64, 1024/32, BATCH), 32, 0, stream>>>(
      (const u32x4*)Wqkv, (const u32x4*)xp, qkv0, sqkv, bqkv, nullptr, 1024, 256, HWX, 0);

  // 3. depthwise 3x3 + BN + ReLU -> branch; swizzle-pack xn = cov-affine(qkv0)
  ACE_dwconv<<<(BATCH*C2C*HWX+255)/256, 256, 0, stream>>>(qkv0, w_dw, s_dw, b_dw, branch);
  {
    int total = BATCH*512*HWX;
    ACE_pack2s<<<(total+255)/256, 256, 0, stream>>>(qkv0, xnp, cov_s, cov_b, 1024, 32, HWX, total);
  }

  // 4. stripe branch (vertical 7x1)
  ACE_stripe_v<<<dim3(HH, 64/32, BATCH), 32, 0, stream>>>((const u32x4*)kvA, (const u32x4*)xnp, x1, 64, 1024, 0);
  ACE_actdn_stats<<<512, 256, 0, stream>>>(x1, stats1);
  ACE_actdn_apply<<<(BATCH*8*HWX+255)/256, 256, 0, stream>>>(x1, stats1, x1sp);
  ACE_stripe_v<<<dim3(HH, 1024/32, BATCH), 32, 0, stream>>>((const u32x4*)kvB, (const u32x4*)x1sp, branch, 1024, 64, 1);
  //    stripe branch (horizontal 1x7)
  ACE_stripe_h<<<dim3(HH, 64/32, BATCH), 32, 0, stream>>>((const u32x4*)kv3A, (const u32x4*)xnp, x3, 64, 1024, 0);
  ACE_actdn_stats<<<512, 256, 0, stream>>>(x3, stats3);
  ACE_actdn_apply<<<(BATCH*8*HWX+255)/256, 256, 0, stream>>>(x3, stats3, x3sp);
  ACE_stripe_h<<<dim3(HH, 1024/32, BATCH), 32, 0, stream>>>((const u32x4*)kv3B, (const u32x4*)x3sp, branch, 1024, 64, 1);

  // 5. swizzle-pack branch; pointwise conv + BN -> qkv_pw
  {
    int total = BATCH*512*HWX;
    ACE_pack2s<<<(total+255)/256, 256, 0, stream>>>(branch, branchp, nullptr, nullptr, 1024, 32, HWX, total);
  }
  ACE_gemm1x1<<<dim3(HWX/64, 256/32, BATCH), 32, 0, stream>>>(
      (const u32x4*)Wpw, (const u32x4*)branchp, qkv_pw, s_pw, b_pw, nullptr, 256, 1024, HWX, 0);

  // 6. row/col attention path
  ACE_mean_w<<<(BATCH*C2C*64+255)/256, 256, 0, stream>>>(qkv0, meanW);
  ACE_mean_h<<<(BATCH*C2C*64+255)/256, 256, 0, stream>>>(qkv0, meanH);
  ACE_attn_axis<<<BATCH*8, 64, 0, stream>>>(meanW, pe_rowq, pe_rowk, out_rp);
  ACE_attn_axis<<<BATCH*8, 64, 0, stream>>>(meanH, pe_colq, pe_colk, out_cp);
  ACE_gemm1x1<<<dim3(1, 512/32, BATCH), 32, 0, stream>>>(
      (const u32x4*)Wrow, (const u32x4*)out_rp, out_rc, s_row, b_row, nullptr, 512, 512, 64, 0);
  ACE_gemm1x1<<<dim3(1, 512/32, BATCH), 32, 0, stream>>>(
      (const u32x4*)Wcol, (const u32x4*)out_cp, out_cc, s_col, b_col, nullptr, 512, 512, 64, 0);

  // 7. v + broadcasts, ReLU (swizzled packed), proj conv + BN, sigmoid-gate -> d_out
  ACE_vsum<<<(BATCH*16*HWX*16+255)/256, 256, 0, stream>>>(qkv0, out_rc, out_cc, vsump);
  ACE_gemm1x1<<<dim3(HWX/64, 256/32, BATCH), 32, 0, stream>>>(
      (const u32x4*)Wproj, (const u32x4*)vsump, out, s_proj, b_proj, qkv_pw, 256, 512, HWX, 1);
}
